// LSTMLayer_28252294873850
// MI455X (gfx1250) — matrix operand order repaired and checked by reading, not compiled, not run on hardware
//
#include <hip/hip_runtime.h>

typedef __attribute__((ext_vector_type(16))) _Float16 v16h;
typedef __attribute__((ext_vector_type(8)))  float    v8f;
typedef __attribute__((ext_vector_type(4)))  unsigned int u32x4;
typedef __attribute__((ext_vector_type(8)))  int      i32x8;
typedef __attribute__((ext_vector_type(4)))  int      i32x4;

#define B_    64
#define T_    512
#define I_    512
#define H_    512
#define NG_   2048      // 4*H gate columns
#define NWG   64        // persistent workgroups (one per WGP)
#define COLS  8         // h-columns owned per recurrence WG
#define GCOLS 32        // gate columns per recurrence WG (4 gates * 8)
#define PADK  520       // padded f16 K-stride (1040B = 260 dw; 260%64=4 -> conflict-free b128)

#if defined(__has_builtin)
#if __has_builtin(__builtin_amdgcn_tensor_load_to_lds) && __has_builtin(__builtin_amdgcn_s_wait_tensorcnt)
#define HAVE_TDM 1
#endif
#endif

#if defined(HAVE_TDM)
// Issue one 2D TDM tile load: rows x rowElems f16, contiguous rows in global,
// into LDS at lds_off with 16B pad per 1024B row (-> PADK stride in LDS).
// D# layout per CDNA5 ISA sec 8.3/8.4.
__device__ __forceinline__ void tdm_load_2d(unsigned lds_off, const void* gptr,
                                            unsigned rows, unsigned rowElems) {
    unsigned long long ga = (unsigned long long)(uintptr_t)gptr;
    u32x4 g0;
    g0.x = 1u;                                                // count=1, user mode
    g0.y = lds_off;                                           // lds_addr (bytes)
    g0.z = (unsigned)(ga & 0xFFFFFFFFull);                    // global_addr[31:0]
    g0.w = (unsigned)((ga >> 32) & 0x1FFFFFFull) | (2u << 30);// global_addr[56:32] | type=2
    i32x8 g1;
    g1[0] = (1 << 16) | (1 << 20) | (7 << 22) | (3 << 25);    // data_size=2B, pad_en, ivl=256dw, amt=4dw
    g1[1] = (int)((rowElems & 0xFFFFu) << 16);                // tensor_dim0[15:0]
    g1[2] = (int)((rowElems >> 16) | ((rows & 0xFFFFu) << 16)); // dim0[31:16] | dim1[15:0]
    g1[3] = (int)((rows >> 16) | ((rowElems & 0xFFFFu) << 16)); // dim1[31:16] | tile_dim0
    g1[4] = (int)rows;                                        // tile_dim1 (tile_dim2=0)
    g1[5] = (int)rowElems;                                    // tensor_dim0_stride[31:0]
    g1[6] = 0; g1[7] = 0;
    i32x4 z4 = {0, 0, 0, 0};
#if __clang_major__ >= 23
    i32x8 z8 = {0, 0, 0, 0, 0, 0, 0, 0};
    __builtin_amdgcn_tensor_load_to_lds(g0, g1, z4, z4, z8, 0);
#else
    __builtin_amdgcn_tensor_load_to_lds(g0, g1, z4, z4, 0);
#endif
}
#endif

// ---------------- prep kernels (parallel, off critical path) ----------------

__global__ void pack_w_kernel(const float* __restrict__ Wf, const float* __restrict__ Wi,
                              const float* __restrict__ Wc, const float* __restrict__ Wo,
                              _Float16* __restrict__ Wxpack, _Float16* __restrict__ Whpack) {
    int idx = blockIdx.x * blockDim.x + threadIdx.x;   // 4 * 1024 * 512
    if (idx >= 4 * 1024 * H_) return;
    int g   = idx / (1024 * H_);
    int rem = idx % (1024 * H_);
    int k   = rem / H_;
    int col = rem % H_;
    const float* W = (g == 0) ? Wf : (g == 1) ? Wi : (g == 2) ? Wc : Wo;
    _Float16 v = (_Float16)W[(size_t)k * H_ + col];
    size_t gcol = (size_t)g * H_ + col;                // [2048] x [512], K contiguous
    if (k < I_) Wxpack[gcol * I_ + k] = v;
    else        Whpack[gcol * H_ + (k - I_)] = v;
}

__global__ void pack_x_kernel(const float* __restrict__ x, _Float16* __restrict__ Xh) {
    size_t idx = (size_t)blockIdx.x * blockDim.x + threadIdx.x; // B*T*I = 16M
    if (idx >= (size_t)B_ * T_ * I_) return;
    int b = (int)(idx / ((size_t)T_ * I_));
    int r = (int)(idx % ((size_t)T_ * I_));
    int t = r / I_;
    int i = r % I_;
    Xh[((size_t)t * B_ + b) * I_ + i] = (_Float16)x[idx];   // [T][B][I] : GEMM row = t*64+b
}

__global__ void prep_misc_kernel(const float* __restrict__ bf, const float* __restrict__ bi,
                                 const float* __restrict__ bc, const float* __restrict__ bo,
                                 float* __restrict__ biasPack, _Float16* __restrict__ hbuf,
                                 unsigned* __restrict__ syncp) {
    int idx = blockIdx.x * blockDim.x + threadIdx.x;
    if (idx < 4 * H_) {
        int g = idx / H_, c = idx % H_;
        const float* b = (g == 0) ? bf : (g == 1) ? bi : (g == 2) ? bc : bo;
        biasPack[idx] = b[c];
    }
    if (idx < 2 * B_ * H_) hbuf[idx] = (_Float16)0.f;      // h0 = 0, both buffers
    if (idx == 0) *syncp = 0u;                             // deterministic per launch
}

// ------------- phase 1: full-chip GEMM  G = Xh @ Wx + bias (permuted store) -------------
// M = B*T = 32768, N = 2048, K = 512. 128x128 tile per WG, single LDS stage (K fits).

__global__ void __launch_bounds__(256, 1)
xw_gemm_kernel(const _Float16* __restrict__ Xh, const _Float16* __restrict__ Wxpack,
               const float* __restrict__ biasPack, float* __restrict__ Gp) {
    extern __shared__ char smem[];
    _Float16* Alds = (_Float16*)smem;                        // [128][PADK]
    _Float16* Blds = (_Float16*)(smem + 128 * PADK * 2);     // [128][PADK] (col-major slice)

    const int bm = blockIdx.x;          // 256 M-blocks
    const int bn = blockIdx.y;          // 16  N-blocks
    const int tid = threadIdx.x;
    const int lane = tid & 31;
    const int wv = tid >> 5;
    const int wm = wv & 3;              // M-tile pair {2wm, 2wm+1}
    const int wn = wv >> 2;             // N-tile quad {4wn .. 4wn+3}

    const _Float16* Asrc = Xh + (size_t)bm * 128 * I_;            // 128 contiguous rows
    const _Float16* Bsrc = Wxpack + (size_t)bn * 128 * I_;        // 128 contiguous cols

#if defined(HAVE_TDM)
    if (wv == 0) {
        tdm_load_2d((unsigned)(uintptr_t)(void*)Alds, Asrc, 128u, (unsigned)I_);
        tdm_load_2d((unsigned)(uintptr_t)(void*)Blds, Bsrc, 128u, (unsigned)I_);
    }
    __builtin_amdgcn_s_wait_tensorcnt((unsigned short)0);
#else
    for (int idx = tid; idx < 128 * (I_ / 8); idx += 256) {
        int r = idx >> 6, seg = idx & 63;
        *(u32x4*)(Alds + r * PADK + seg * 8) = *(const u32x4*)(Asrc + (size_t)r * I_ + seg * 8);
        *(u32x4*)(Blds + r * PADK + seg * 8) = *(const u32x4*)(Bsrc + (size_t)r * I_ + seg * 8);
    }
#endif
    __syncthreads();

    const int asel = (lane < 16) ? 0 : 16;
    const int bsel = (lane < 16) ? 0 : 16;
    const int l15  = lane & 15;

    v8f acc[2][4];
    #pragma unroll
    for (int i = 0; i < 2; ++i)
        #pragma unroll
        for (int j = 0; j < 4; ++j) acc[i][j] = (v8f){};

    for (int k0 = 0; k0 < I_; k0 += 32) {
        union { v16h v; u32x4 q[2]; } af[2], bf[4];
        #pragma unroll
        for (int i = 0; i < 2; ++i) {
            const _Float16* ap = Alds + ((wm * 2 + i) * 16 + l15) * PADK + k0 + asel;
            af[i].q[0] = *(const u32x4*)(ap);
            af[i].q[1] = *(const u32x4*)(ap + 8);
        }
        #pragma unroll
        for (int j = 0; j < 4; ++j) {
            const _Float16* bp = Blds + ((wn * 4 + j) * 16 + l15) * PADK + k0 + bsel;
            bf[j].q[0] = *(const u32x4*)(bp);
            bf[j].q[1] = *(const u32x4*)(bp + 8);
        }
        #pragma unroll
        for (int i = 0; i < 2; ++i)
            #pragma unroll
            for (int j = 0; j < 4; ++j)
                acc[i][j] = __builtin_amdgcn_wmma_f32_16x16x32_f16(
                    false, af[i].v, false, bf[j].v, (short)0, acc[i][j], false, false);
    }

    // epilogue: bias add + permuted store Gp[wg][t][b][gate*8+hc]
    const int mlane = (lane < 16) ? 0 : 8;
    #pragma unroll
    for (int i = 0; i < 2; ++i) {
        #pragma unroll
        for (int j = 0; j < 4; ++j) {
            int gcol = bn * 128 + (wn * 4 + j) * 16 + l15;
            int gate = gcol >> 9, hcol = gcol & 511;
            int wgi  = hcol >> 3, lc = (gate << 3) | (hcol & 7);
            float bias = biasPack[gcol];
            #pragma unroll
            for (int r = 0; r < 8; ++r) {
                int gr = bm * 128 + (wm * 2 + i) * 16 + mlane + r;  // = t*64 + b
                int t = gr >> 6, b = gr & 63;
                Gp[(((size_t)wgi * T_ + t) * B_ + b) * GCOLS + lc] = acc[i][j][r] + bias;
            }
        }
    }
}

// ------------- phase 2: persistent recurrence, K = 512 (h-part only) -------------

__global__ void __launch_bounds__(256, 1)
lstm_persistent(const _Float16* __restrict__ Whpack, const float* __restrict__ Gp,
                _Float16* __restrict__ hbuf, unsigned* __restrict__ syncp,
                float* __restrict__ out) {
    extern __shared__ char smem[];
    _Float16* comb = (_Float16*)smem;                             // [64][PADK]  h_prev tile
    _Float16* wlds = (_Float16*)(smem + 64 * PADK * 2);           // [32][PADK]  Wh slice
    float*    gate = (float*)(smem + (64 + 32) * PADK * 2);       // [64][32]

    const int wg   = blockIdx.x;
    const int tid  = threadIdx.x;
    const int lane = tid & 31;
    const int wv   = tid >> 5;
    const int m0   = (wv & 3) * 16;
    const int n0   = (wv >> 2) * 16;

    // resident Wh slice: 4 gates x 8 cols, K=512 each, col-major padded
    for (int idx = tid; idx < GCOLS * (H_ / 8); idx += 256) {
        int c = idx >> 6, seg = idx & 63;
        int g = c >> 3, hc = c & 7;
        *(u32x4*)(wlds + c * PADK + seg * 8) =
            *(const u32x4*)(Whpack + ((size_t)(g * H_ + wg * COLS + hc)) * H_ + seg * 8);
    }

    float cst0 = 0.f, cst1 = 0.f;

    const int asel  = (lane < 16) ? 0 : 16;
    const int arow  = m0 + (lane & 15);
    const int bcol  = n0 + (lane & 15);
    const int bsel  = (lane < 16) ? 0 : 16;
    const int mbase = m0 + ((lane < 16) ? 0 : 8);
    const int ccol  = n0 + (lane & 15);

    // prefetch x-part slice for t=0 (speculative, L2-resident after GEMM)
    __builtin_prefetch(Gp + (size_t)wg * T_ * B_ * GCOLS + tid * 32, 0, 1);

    for (int t = 0; t < T_; ++t) {
        const _Float16* hread  = hbuf + (size_t)(t & 1) * B_ * H_;
        _Float16*       hwrite = hbuf + (size_t)((t + 1) & 1) * B_ * H_;

        // ---- stage h_prev tile [64][512] f16 via TDM (pad -> PADK stride) ----
#if defined(HAVE_TDM)
        if (wv == 0)
            tdm_load_2d((unsigned)(uintptr_t)(void*)comb, hread, (unsigned)B_, (unsigned)H_);
#else
        for (int idx = tid; idx < B_ * (H_ / 8); idx += 256) {
            int row = idx >> 6, seg = idx & 63;
            *(u32x4*)(comb + row * PADK + seg * 8) =
                *(const u32x4*)(hread + (size_t)row * H_ + seg * 8);
        }
#endif
        // ---- init accumulators from precomputed x-part (overlaps the DMA) ----
        v8f acc;
        {
            const float* gsl = Gp + ((size_t)wg * T_ + t) * B_ * GCOLS;
            #pragma unroll
            for (int r = 0; r < 8; ++r)
                acc[r] = gsl[(mbase + r) * GCOLS + ccol];
        }
#if defined(HAVE_TDM)
        __builtin_amdgcn_s_wait_tensorcnt((unsigned short)0);
#endif
        __syncthreads();

        // ---- recurrent GEMM: 16 chained WMMAs (K = 512) ----
        for (int k0 = 0; k0 < H_; k0 += 32) {
            union { v16h v; u32x4 q[2]; } af, bfg;
            const _Float16* ap = comb + arow * PADK + k0 + asel;
            af.q[0]  = *(const u32x4*)(ap);
            af.q[1]  = *(const u32x4*)(ap + 8);
            const _Float16* bp = wlds + bcol * PADK + k0 + bsel;
            bfg.q[0] = *(const u32x4*)(bp);
            bfg.q[1] = *(const u32x4*)(bp + 8);
            acc = __builtin_amdgcn_wmma_f32_16x16x32_f16(false, af.v, false, bfg.v,
                                                         (short)0, acc, false, false);
        }
        #pragma unroll
        for (int r = 0; r < 8; ++r)
            gate[(mbase + r) * GCOLS + ccol] = acc[r];

        // ---- prefetch next step's x-part slice into cache (hides L2 latency
        //      behind the elementwise phase and the grid sync) ----
        if (t + 1 < T_)
            __builtin_prefetch(Gp + ((size_t)wg * T_ + (t + 1)) * B_ * GCOLS + tid * 32, 0, 1);

        __syncthreads();

        // ---- fused nonlinearity + cell update (bias already folded into Gp) ----
        #pragma unroll
        for (int rep = 0; rep < 2; ++rep) {
            int p  = tid + rep * 256;
            int b  = p >> 3, hc = p & 7;
            float fp = gate[b * GCOLS +      hc];
            float ip = gate[b * GCOLS +  8 + hc];
            float gp = gate[b * GCOLS + 16 + hc];
            float op = gate[b * GCOLS + 24 + hc];
            float ft = 1.f / (1.f + __expf(-fp));
            float it = 1.f / (1.f + __expf(-ip));
            float ct = 2.f / (1.f + __expf(-2.f * gp)) - 1.f;   // tanh
            float ot = 1.f / (1.f + __expf(-op));
            float cs = rep ? cst1 : cst0;
            cs = ft * cs + it * ct;
            if (rep) cst1 = cs; else cst0 = cs;
            float th = 2.f / (1.f + __expf(-2.f * cs)) - 1.f;
            float ht = ot * th;
            int ghc = wg * COLS + hc;
            out[((size_t)b * T_ + t) * H_ + ghc] = ht;
            hwrite[(size_t)b * H_ + ghc] = (_Float16)ht;
        }

        // ---- device-wide sync: monotonic counter ----
        __syncthreads();
        if (tid == 0) {
            __threadfence();
            atomicAdd(syncp, 1u);
            unsigned target = (unsigned)NWG * (unsigned)(t + 1);
            while (__hip_atomic_load(syncp, __ATOMIC_ACQUIRE, __HIP_MEMORY_SCOPE_AGENT) < target)
                __builtin_amdgcn_s_sleep(1);
        }
        __syncthreads();
    }
}

// ---------------- host entry ----------------

extern "C" void kernel_launch(void* const* d_in, const int* in_sizes, int n_in,
                              void* d_out, int out_size, void* d_ws, size_t ws_size,
                              hipStream_t stream) {
    const float* x  = (const float*)d_in[0];
    const float* Wf = (const float*)d_in[1];
    const float* bf = (const float*)d_in[2];
    const float* Wi = (const float*)d_in[3];
    const float* bi = (const float*)d_in[4];
    const float* Wc = (const float*)d_in[5];
    const float* bc = (const float*)d_in[6];
    const float* Wo = (const float*)d_in[7];
    const float* bo = (const float*)d_in[8];

    char* ws = (char*)d_ws;
    _Float16* Wxpack = (_Float16*)ws;   ws += (size_t)NG_ * I_ * 2;      // 2 MB
    _Float16* Whpack = (_Float16*)ws;   ws += (size_t)NG_ * H_ * 2;      // 2 MB
    float*    biasPk = (float*)ws;      ws += (size_t)NG_ * 4;           // 8 KB
    _Float16* hbuf   = (_Float16*)ws;   ws += (size_t)2 * B_ * H_ * 2;   // 128 KB
    unsigned* syncp  = (unsigned*)ws;   ws += 256;
    _Float16* Xh     = (_Float16*)ws;   ws += (size_t)T_ * B_ * I_ * 2;  // 32 MB
    float*    Gp     = (float*)ws;                                        // 256 MB

    pack_w_kernel<<<(4 * 1024 * H_ + 255) / 256, 256, 0, stream>>>(Wf, Wi, Wc, Wo, Wxpack, Whpack);
    prep_misc_kernel<<<(2 * B_ * H_ + 255) / 256, 256, 0, stream>>>(bf, bi, bc, bo,
                                                                    biasPk, hbuf, syncp);
    pack_x_kernel<<<(int)(((size_t)B_ * T_ * I_ + 255) / 256), 256, 0, stream>>>(x, Xh);

    size_t smemG = (size_t)2 * 128 * PADK * 2;                       // ~266 KB
    xw_gemm_kernel<<<dim3(256, 16), 256, smemG, stream>>>(Xh, Wxpack, biasPk, Gp);

    size_t smemR = (size_t)(64 + 32) * PADK * 2 + 64 * 32 * 4;       // ~108 KB
    lstm_persistent<<<NWG, 256, smemR, stream>>>(Whpack, Gp, hbuf, syncp, (float*)d_out);
}
